// FastSpeech_87205015978521
// MI455X (gfx1250) — matrix operand order repaired and checked by reading, not compiled, not run on hardware
//
#include <hip/hip_runtime.h>
#include <hip/hip_bf16.h>

// ---------------------------------------------------------------------------
// FastSpeech-like forward for MI455X (gfx1250, wave32).
// All dense GEMMs run through v_wmma_f32_16x16x32_f16 (f16 in, f32 accum).
// Fragment loads go straight global->VGPR using one contiguous lane
// k order (A and B: one 16-half run at
// k+16*hi), matching row-major [M,K] activations and torch [N,K] weights with
// zero transposition. Weights are padded to N%64==0 rows at f16-conversion
// time so the K-loop has NO lane guards: EXEC stays all-ones and the inner
// loop is pure b128-load clauses + wmma. Epilogues fuse bias/ReLU/f16-cvt.
// ---------------------------------------------------------------------------

typedef _Float16 half_t;
typedef __attribute__((ext_vector_type(16))) _Float16 v16h;
typedef __attribute__((ext_vector_type(8)))  _Float16 v8h;
typedef __attribute__((ext_vector_type(8)))  float    v8f;

#define NHEAD 2
#define HDIM  256
#define DH    128
#define DFFN  2048
#define BB    16
#define SS    256
#define TT    2048
#define NMEL  80
#define NMELP 128   // NMEL padded to multiple of 64 for guard-free B loads
#define EPSLN 1e-5f
#define LN1E4_OVER_H 0.035977892078031965f  // ln(10000)/256

__device__ __forceinline__ float pe_val(int pos, int h) {
    // pe[pos, 2i]   = sin(pos * exp(-2i ln(1e4)/E))
    // pe[pos, 2i+1] = cos(pos * exp(-2i ln(1e4)/E))
    float den = expf(-(float)(h & ~1) * LN1E4_OVER_H);
    float a = (float)pos * den;
    return (h & 1) ? cosf(a) : sinf(a);
}

// ---------------------- f32 -> f16 with zero tail pad ----------------------
__global__ void f32_to_f16_pad_kernel(const float* __restrict__ src,
                                      half_t* __restrict__ dst,
                                      int n_src, int n_tot) {
    int i = blockIdx.x * blockDim.x + threadIdx.x;
    if (i < n_tot) dst[i] = (i < n_src) ? (half_t)src[i] : (half_t)0.f;
}

// --------------------------- embedding + pos-enc ---------------------------
__global__ __launch_bounds__(256) void embed_pe_kernel(
    const int* __restrict__ tokens, const float* __restrict__ embed,
    float* __restrict__ x32, half_t* __restrict__ x16) {
    int row = blockIdx.x;            // b*S + s
    int h = threadIdx.x;             // 0..255
    int b = row / SS;                // faithful quirk: PE indexed by batch
    int tok = tokens[row];
    float v = embed[(size_t)tok * HDIM + h] + pe_val(b, h);
    x32[(size_t)row * HDIM + h] = v;
    x16[(size_t)row * HDIM + h] = (half_t)v;
}

// ------------------------------- WMMA GEMM ---------------------------------
// C[M,N] = A[M,K] @ W[Npad,K]^T + bias ; optional ReLU; f32 (Cf) or f16 (Ch).
// W must have ceil(N/64)*64 valid rows (zero-padded). One wave -> 16x64 strip
// (A frag reused over 4 N tiles); block = 4 waves -> 64x64 tile.
__global__ __launch_bounds__(128) void gemm_f16_wmma(
    const half_t* __restrict__ A, const half_t* __restrict__ W,
    const float* __restrict__ bias,
    float* __restrict__ Cf, half_t* __restrict__ Ch,
    int M, int N, int K, int relu)
{
    const int lane  = threadIdx.x & 31;
    const int wave  = threadIdx.x >> 5;
    const int mtile = blockIdx.y * 4 + wave;
    const int m0    = mtile * 16;
    if (m0 >= M) return;
    const int nbase = blockIdx.x * 64;

    const int lrow = lane & 15;   // M (A frag) / N (B frag) within tile
    const int lhi  = lane >> 4;   // lane half-group

    // A 16x32 f16 layout: halves 0..15 -> K = k0 + 16*lhi + {0..15}
    const half_t* arow = A + (size_t)(m0 + lrow) * K + 16 * lhi;
    // B 32x16 f16 layout: halves 0..15 -> K = k0 + 16*lhi + {0..15}
    const half_t* brow0 = W + (size_t)(nbase +  0 + lrow) * K + 16 * lhi;
    const half_t* brow1 = W + (size_t)(nbase + 16 + lrow) * K + 16 * lhi;
    const half_t* brow2 = W + (size_t)(nbase + 32 + lrow) * K + 16 * lhi;
    const half_t* brow3 = W + (size_t)(nbase + 48 + lrow) * K + 16 * lhi;

    v8f zero8 = {};
    v8f acc0 = zero8, acc1 = zero8, acc2 = zero8, acc3 = zero8;

    for (int k0 = 0; k0 < K; k0 += 32) {
        v8h alo = *(const v8h*)(arow + k0);
        v8h ahi = *(const v8h*)(arow + k0 + 8);
        v16h afrag = __builtin_shufflevector(
            alo, ahi, 0, 1, 2, 3, 4, 5, 6, 7, 8, 9, 10, 11, 12, 13, 14, 15);
        __builtin_prefetch(arow + k0 + 128, 0, 3);  // global_prefetch_b8

        v8h b0lo = *(const v8h*)(brow0 + k0);
        v8h b0hi = *(const v8h*)(brow0 + k0 + 8);
        v8h b1lo = *(const v8h*)(brow1 + k0);
        v8h b1hi = *(const v8h*)(brow1 + k0 + 8);
        v8h b2lo = *(const v8h*)(brow2 + k0);
        v8h b2hi = *(const v8h*)(brow2 + k0 + 8);
        v8h b3lo = *(const v8h*)(brow3 + k0);
        v8h b3hi = *(const v8h*)(brow3 + k0 + 8);

        v16h bf0 = __builtin_shufflevector(
            b0lo, b0hi, 0, 1, 2, 3, 4, 5, 6, 7, 8, 9, 10, 11, 12, 13, 14, 15);
        v16h bf1 = __builtin_shufflevector(
            b1lo, b1hi, 0, 1, 2, 3, 4, 5, 6, 7, 8, 9, 10, 11, 12, 13, 14, 15);
        v16h bf2 = __builtin_shufflevector(
            b2lo, b2hi, 0, 1, 2, 3, 4, 5, 6, 7, 8, 9, 10, 11, 12, 13, 14, 15);
        v16h bf3 = __builtin_shufflevector(
            b3lo, b3hi, 0, 1, 2, 3, 4, 5, 6, 7, 8, 9, 10, 11, 12, 13, 14, 15);

        acc0 = __builtin_amdgcn_wmma_f32_16x16x32_f16(
            false, afrag, false, bf0, (short)0, acc0, false, false);
        acc1 = __builtin_amdgcn_wmma_f32_16x16x32_f16(
            false, afrag, false, bf1, (short)0, acc1, false, false);
        acc2 = __builtin_amdgcn_wmma_f32_16x16x32_f16(
            false, afrag, false, bf2, (short)0, acc2, false, false);
        acc3 = __builtin_amdgcn_wmma_f32_16x16x32_f16(
            false, afrag, false, bf3, (short)0, acc3, false, false);
    }

    // C/D layout: VGPR r, lane -> M = m0 + r + 8*lhi, N = ntile*16 + lrow
    v8f accs[4] = { acc0, acc1, acc2, acc3 };
#pragma unroll
    for (int t = 0; t < 4; ++t) {
        int n = nbase + t * 16 + lrow;
        if (n >= N) continue;                       // only final GEMM tail
        float bv = bias ? bias[n] : 0.f;
#pragma unroll
        for (int r = 0; r < 8; ++r) {
            int m = m0 + r + 8 * lhi;
            float v = accs[t][r] + bv;
            if (relu) v = fmaxf(v, 0.f);
            if (Cf) Cf[(size_t)m * N + n] = v;
            else    Ch[(size_t)m * N + n] = (half_t)v;
        }
    }
}

// -------------------------- tiny 16x16 attention ---------------------------
// qkv: [l*Nax + n, 3H] f16 ; attention mixes the 16 axis-0 entries per (n,h).
__global__ __launch_bounds__(256) void attention_kernel(
    const half_t* __restrict__ qkv, half_t* __restrict__ out, int Nax)
{
    __shared__ float qs[16][DH];
    __shared__ float ks[16][DH];
    __shared__ float vs[16][DH];
    __shared__ float sc[16][17];
    const int n = blockIdx.x;
    const int h = blockIdx.y;
    const int tid = threadIdx.x;

    for (int j = 0; j < 8; ++j) {          // 16*128 elems / 256 threads
        int idx = tid + 256 * j;
        int l = idx >> 7, d = idx & 127;
        size_t base = ((size_t)(l * Nax + n)) * (3 * HDIM) + h * DH + d;
        qs[l][d] = (float)qkv[base];
        ks[l][d] = (float)qkv[base + HDIM];
        vs[l][d] = (float)qkv[base + 2 * HDIM];
    }
    __syncthreads();
    {
        int l = tid >> 4, m = tid & 15;
        float a = 0.f;
        for (int d = 0; d < DH; ++d) a += qs[l][d] * ks[m][d];
        sc[l][m] = a * 0.08838834764831845f;   // 1/sqrt(128)
    }
    __syncthreads();
    if (tid < 16) {
        int l = tid;
        float mx = sc[l][0];
        for (int m = 1; m < 16; ++m) mx = fmaxf(mx, sc[l][m]);
        float s = 0.f;
        for (int m = 0; m < 16; ++m) { float e = expf(sc[l][m] - mx); sc[l][m] = e; s += e; }
        float inv = 1.f / s;
        for (int m = 0; m < 16; ++m) sc[l][m] *= inv;
    }
    __syncthreads();
    for (int j = 0; j < 8; ++j) {
        int idx = tid + 256 * j;
        int l = idx >> 7, d = idx & 127;
        float a = 0.f;
        for (int m = 0; m < 16; ++m) a += sc[l][m] * vs[m][d];
        out[((size_t)(l * Nax + n)) * HDIM + h * DH + d] = (half_t)a;
    }
}

// ----------------------- (residual +) LayerNorm (+relu) --------------------
__global__ __launch_bounds__(256) void ln_kernel(
    const float* __restrict__ X, const float* __restrict__ Yres,
    const float* __restrict__ g, const float* __restrict__ bta,
    float* __restrict__ out32, half_t* __restrict__ out16, int relu)
{
    __shared__ float red[256];
    const int row = blockIdx.x, h = threadIdx.x;
    float x = X[(size_t)row * HDIM + h];
    if (Yres) x += Yres[(size_t)row * HDIM + h];
    red[h] = x; __syncthreads();
    for (int s = 128; s > 0; s >>= 1) { if (h < s) red[h] += red[h + s]; __syncthreads(); }
    float mu = red[0] * (1.f / HDIM);
    __syncthreads();
    float c = x - mu;
    red[h] = c * c; __syncthreads();
    for (int s = 128; s > 0; s >>= 1) { if (h < s) red[h] += red[h + s]; __syncthreads(); }
    float var = red[0] * (1.f / HDIM);
    float v = c * rsqrtf(var + EPSLN) * g[h] + bta[h];
    if (relu) v = fmaxf(v, 0.f);
    out32[(size_t)row * HDIM + h] = v;
    if (out16) out16[(size_t)row * HDIM + h] = (half_t)v;
}

// ------------------------------ Conv1d 'same' ------------------------------
// y[b,s,o] = sum_{i,k} x[b, s+k-5, i] * w[o,i,k] + bias[o], K=11
__global__ __launch_bounds__(256) void conv1d_kernel(
    const float* __restrict__ X, const float* __restrict__ W,
    const float* __restrict__ bias, float* __restrict__ out)
{
    __shared__ float xs[11][HDIM];
    const int row = blockIdx.x;      // b*S + s
    const int b = row / SS, s = row % SS;
    const int o = threadIdx.x;
    for (int k = 0; k < 11; ++k) {
        int sp = s + k - 5;
        xs[k][o] = (sp >= 0 && sp < SS) ? X[((size_t)(b * SS + sp)) * HDIM + o] : 0.f;
    }
    __syncthreads();
    float acc = bias[o];
    for (int k = 0; k < 11; ++k) {
        const float* wr = W + (size_t)o * HDIM * 11 + k;
        for (int i = 0; i < HDIM; ++i) acc += xs[k][i] * wr[(size_t)i * 11];
    }
    out[(size_t)row * HDIM + o] = acc;
}

// -------------------- duration head: dot + mask by length ------------------
__global__ __launch_bounds__(256) void lin_mask_kernel(
    const float* __restrict__ X, const float* __restrict__ lw,
    const float* __restrict__ lb, const int* __restrict__ token_nums,
    float* __restrict__ mout)
{
    __shared__ float red[256];
    const int row = blockIdx.x, h = threadIdx.x;
    red[h] = X[(size_t)row * HDIM + h] * lw[h];
    __syncthreads();
    for (int s = 128; s > 0; s >>= 1) { if (h < s) red[h] += red[h + s]; __syncthreads(); }
    if (h == 0) {
        int b = row / SS, s = row % SS;
        float v = red[0] + lb[0];
        mout[row] = (s < token_nums[b]) ? v : 0.f;
    }
}

// ------------------------ length regulator: indices ------------------------
__global__ __launch_bounds__(256) void expand_idx_kernel(
    const float* __restrict__ align, const int* __restrict__ gt,
    int* __restrict__ idxout)
{
    __shared__ float bd[SS];
    const int b = blockIdx.x, tid = threadIdx.x;
    if (tid == 0) {
        float c = 0.f, g = (float)gt[b];
        for (int s = 0; s < SS; ++s) { c += align[b * SS + s]; bd[s] = fminf(c, g); }
    }
    __syncthreads();
    float last = bd[SS - 1];
    for (int j = 0; j < TT / 256; ++j) {
        int t = tid + 256 * j;
        float tf = (float)t;
        int lo = 0, hi = SS;                 // searchsorted(bd, t, 'right')
        while (lo < hi) { int mid = (lo + hi) >> 1; if (bd[mid] <= tf) lo = mid + 1; else hi = mid; }
        int idx = lo < SS ? lo : SS - 1;
        idxout[b * TT + t] = (tf < last) ? idx : -1;
    }
}

// -------------------- length regulator: gather + pos-enc -------------------
__global__ __launch_bounds__(256) void expand_gather_kernel(
    const float* __restrict__ XE, const int* __restrict__ idx,
    float* __restrict__ XD32, half_t* __restrict__ XD16)
{
    const int row = blockIdx.x;      // b*T + t
    const int h = threadIdx.x;
    const int b = row / TT;
    int id = idx[row];
    float v = (id >= 0) ? XE[((size_t)(b * SS + id)) * HDIM + h] : 0.f;
    v += pe_val(b, h);
    XD32[(size_t)row * HDIM + h] = v;
    XD16[(size_t)row * HDIM + h] = (half_t)v;
}

// ---------------------------------------------------------------------------
extern "C" void kernel_launch(void* const* d_in, const int* in_sizes, int n_in,
                              void* d_out, int out_size, void* d_ws, size_t ws_size,
                              hipStream_t stream)
{
    (void)in_sizes; (void)n_in; (void)out_size; (void)ws_size;

    const int*   tokens     = (const int*)d_in[0];
    const int*   token_nums = (const int*)d_in[1];
    const float* align      = (const float*)d_in[2];
    const int*   gt         = (const int*)d_in[3];
    /* d_in[4] = T_mel scalar (2048, compile-time here) */
    const float* embed      = (const float*)d_in[5];

    struct Layer {
        const float *in_w, *in_b, *out_w, *out_b, *ln1_g, *ln1_b;
        const float *w1, *b1, *w2, *b2, *ln2_g, *ln2_b;
    };
    auto get_layer = [&](int base) {
        Layer L;
        L.in_w  = (const float*)d_in[base + 0];  L.in_b  = (const float*)d_in[base + 1];
        L.out_w = (const float*)d_in[base + 2];  L.out_b = (const float*)d_in[base + 3];
        L.ln1_g = (const float*)d_in[base + 4];  L.ln1_b = (const float*)d_in[base + 5];
        L.w1    = (const float*)d_in[base + 6];  L.b1    = (const float*)d_in[base + 7];
        L.w2    = (const float*)d_in[base + 8];  L.b2    = (const float*)d_in[base + 9];
        L.ln2_g = (const float*)d_in[base + 10]; L.ln2_b = (const float*)d_in[base + 11];
        return L;
    };
    Layer enc0 = get_layer(6), enc1 = get_layer(18);
    const float* c1_w = (const float*)d_in[30]; const float* c1_b = (const float*)d_in[31];
    const float* lrg1 = (const float*)d_in[32]; const float* lrb1 = (const float*)d_in[33];
    const float* c2_w = (const float*)d_in[34]; const float* c2_b = (const float*)d_in[35];
    const float* lrg2 = (const float*)d_in[36]; const float* lrb2 = (const float*)d_in[37];
    const float* lin_w = (const float*)d_in[38]; const float* lin_b = (const float*)d_in[39];
    Layer dec0 = get_layer(40), dec1 = get_layer(52);
    const float* final_w = (const float*)d_in[64];
    const float* final_b = (const float*)d_in[65];

    // ---------------- workspace carve ----------------
    char* ws = (char*)d_ws;
    size_t off = 0;
    auto alloc = [&](size_t bytes) -> char* {
        off = (off + 255) & ~(size_t)255;
        char* p = ws + off; off += bytes; return p;
    };
    const size_t WPL = 1310720;                     // f16 weight halfs per layer
    half_t* w16  = (half_t*)alloc(sizeof(half_t) * (4 * WPL + (size_t)NMELP * HDIM));
    float*  XE32 = (float*) alloc(4096ull  * HDIM * 4);
    half_t* XE16 = (half_t*)alloc(4096ull  * HDIM * 2);
    float*  XD32 = (float*) alloc(32768ull * HDIM * 4);
    half_t* XD16 = (half_t*)alloc(32768ull * HDIM * 2);
    half_t* QKV  = (half_t*)alloc(32768ull * 3 * HDIM * 2);
    half_t* O16  = (half_t*)alloc(32768ull * HDIM * 2);
    float*  Y32  = (float*) alloc(32768ull * HDIM * 4);
    half_t* FFNH = (half_t*)alloc(8192ull  * DFFN * 2);
    float*  CT1  = (float*) alloc(4096ull  * HDIM * 4);
    float*  CT2  = (float*) alloc(4096ull  * HDIM * 4);
    int*    IDX  = (int*)   alloc(32768ull * 4);

    auto cvt = [&](const float* src, half_t* dst, int n) {
        f32_to_f16_pad_kernel<<<(n + 255) / 256, 256, 0, stream>>>(src, dst, n, n);
    };
    const Layer* Ls[4] = { &enc0, &enc1, &dec0, &dec1 };
    for (int li = 0; li < 4; ++li) {
        half_t* base = w16 + (size_t)li * WPL;
        cvt(Ls[li]->in_w,  base,          3 * HDIM * HDIM);   // [768,256]
        cvt(Ls[li]->out_w, base + 196608, HDIM * HDIM);       // [256,256]
        cvt(Ls[li]->w1,    base + 262144, DFFN * HDIM);       // [2048,256]
        cvt(Ls[li]->w2,    base + 786432, HDIM * DFFN);       // [256,2048]
    }
    half_t* fw16 = w16 + 4 * WPL;                   // [NMELP,256], rows 80.. = 0
    {
        int n_src = NMEL * HDIM, n_tot = NMELP * HDIM;
        f32_to_f16_pad_kernel<<<(n_tot + 255) / 256, 256, 0, stream>>>(
            final_w, fw16, n_src, n_tot);
    }

    auto gemm = [&](const half_t* A, const half_t* W, const float* bias,
                    float* Cf, half_t* Ch, int M, int N, int K, int relu) {
        dim3 grid((N + 63) / 64, (M + 63) / 64);
        gemm_f16_wmma<<<grid, 128, 0, stream>>>(A, W, bias, Cf, Ch, M, N, K, relu);
    };

    // ---------------- encoder ----------------
    embed_pe_kernel<<<4096, 256, 0, stream>>>(tokens, embed, XE32, XE16);
    for (int li = 0; li < 2; ++li) {
        const Layer& L = *Ls[li];
        half_t* wb = w16 + (size_t)li * WPL;
        gemm(XE16, wb, L.in_b, nullptr, QKV, 4096, 3 * HDIM, HDIM, 0);
        attention_kernel<<<dim3(SS, NHEAD), 256, 0, stream>>>(QKV, O16, SS);
        gemm(O16, wb + 196608, L.out_b, Y32, nullptr, 4096, HDIM, HDIM, 0);
        ln_kernel<<<4096, 256, 0, stream>>>(XE32, Y32, L.ln1_g, L.ln1_b, XE32, XE16, 0);
        gemm(XE16, wb + 262144, L.b1, nullptr, FFNH, 4096, DFFN, HDIM, 1);
        gemm(FFNH, wb + 786432, L.b2, Y32, nullptr, 4096, HDIM, DFFN, 0);
        ln_kernel<<<4096, 256, 0, stream>>>(XE32, Y32, L.ln2_g, L.ln2_b, XE32, XE16, 0);
    }

    // ---------------- duration predictor branch ----------------
    conv1d_kernel<<<4096, 256, 0, stream>>>(XE32, c1_w, c1_b, CT1);
    ln_kernel<<<4096, 256, 0, stream>>>(CT1, nullptr, lrg1, lrb1, CT1, nullptr, 1);
    conv1d_kernel<<<4096, 256, 0, stream>>>(CT1, c2_w, c2_b, CT2);
    ln_kernel<<<4096, 256, 0, stream>>>(CT2, nullptr, lrg2, lrb2, CT2, nullptr, 1);
    float* m_out = (float*)d_out + (size_t)BB * TT * NMEL;
    lin_mask_kernel<<<4096, 256, 0, stream>>>(CT2, lin_w, lin_b, token_nums, m_out);

    // ---------------- length regulator ----------------
    expand_idx_kernel<<<BB, 256, 0, stream>>>(align, gt, IDX);
    expand_gather_kernel<<<32768, 256, 0, stream>>>(XE32, IDX, XD32, XD16);

    // ---------------- decoder ----------------
    for (int li = 2; li < 4; ++li) {
        const Layer& L = *Ls[li];
        half_t* wb = w16 + (size_t)li * WPL;
        gemm(XD16, wb, L.in_b, nullptr, QKV, 32768, 3 * HDIM, HDIM, 0);
        attention_kernel<<<dim3(TT, NHEAD), 256, 0, stream>>>(QKV, O16, TT);
        gemm(O16, wb + 196608, L.out_b, Y32, nullptr, 32768, HDIM, HDIM, 0);
        ln_kernel<<<32768, 256, 0, stream>>>(XD32, Y32, L.ln1_g, L.ln1_b, XD32, XD16, 0);
        for (int c = 0; c < 32768; c += 8192) {   // chunk FFN hidden (ws economy)
            gemm(XD16 + (size_t)c * HDIM, wb + 262144, L.b1, nullptr, FFNH, 8192, DFFN, HDIM, 1);
            gemm(FFNH, wb + 786432, L.b2, Y32 + (size_t)c * HDIM, nullptr, 8192, HDIM, DFFN, 0);
        }
        ln_kernel<<<32768, 256, 0, stream>>>(XD32, Y32, L.ln2_g, L.ln2_b, XD32, XD16, 0);
    }

    // ---------------- final projection to mels (W padded to 128 rows) ------
    gemm(XD16, fw16, final_b, (float*)d_out, nullptr, 32768, NMEL, HDIM, 0);
}
